// BigBirdSummaryGenerator_57475252355350
// MI455X (gfx1250) — matrix operand order repaired and checked by reading, not compiled, not run on hardware
//
#include <hip/hip_runtime.h>
#include <hip/hip_bf16.h>

// ---------------- model dims (BigBird-roberta-base, S=2048) ----------------
#define SS   2048
#define DD   768
#define FF   3072
#define VV   50358
#define LL   12
#define HH   12
#define HDIM 64
#define BSZ  64
#define NBLK 32
#define NKBL 8          // 5 fixed + 3 random key blocks per query block
#define NEGBIG (-1.0e9f)

typedef __bf16 bf16_t;
typedef __attribute__((ext_vector_type(16))) __bf16 v16bf;
typedef __attribute__((ext_vector_type(8)))  __bf16 v8bf;
typedef __attribute__((ext_vector_type(4)))  __bf16 v4bf;
typedef __attribute__((ext_vector_type(8)))  float  v8f;

// native conversion: lowers to hardware cvt (RNE), far cheaper than bit-twiddle
__device__ __forceinline__ bf16_t f2bf(float f) { return (bf16_t)f; }
__device__ __forceinline__ v4bf f2bf4(float4 f) {
  v4bf b;
  b[0] = (bf16_t)f.x; b[1] = (bf16_t)f.y; b[2] = (bf16_t)f.z; b[3] = (bf16_t)f.w;
  return b;
}

__device__ __forceinline__ v8f vzero8() {
  v8f z;
#pragma unroll
  for (int i = 0; i < 8; ++i) z[i] = 0.0f;
  return z;
}

// A-operand fragment: two contiguous 8-element runs at p[0..7] and p[16..23]
__device__ __forceinline__ v16bf ld_dual8(const bf16_t* p) {
  v8bf lo = *(const v8bf*)(p);
  v8bf hi = *(const v8bf*)(p + 16);
  return __builtin_shufflevector(lo, hi, 0, 1, 2, 3, 4, 5, 6, 7,
                                 8, 9, 10, 11, 12, 13, 14, 15);
}
// B-operand fragment: one contiguous 16-element run
__device__ __forceinline__ v16bf ld_cont16(const bf16_t* p) {
  return *(const v16bf*)(p);
}

__device__ __forceinline__ float redsum32(float x) {
#pragma unroll
  for (int off = 16; off >= 1; off >>= 1) x += __shfl_xor(x, off, 32);
  return x;
}
__device__ __forceinline__ float redmax16(float x) {   // within 16-lane half
#pragma unroll
  for (int off = 8; off >= 1; off >>= 1) x = fmaxf(x, __shfl_xor(x, off, 32));
  return x;
}
__device__ __forceinline__ float redsum16(float x) {
#pragma unroll
  for (int off = 8; off >= 1; off >>= 1) x += __shfl_xor(x, off, 32);
  return x;
}

__device__ __forceinline__ float gelu_tanh(float x) {
  const float c = 0.7978845608028654f;   // sqrt(2/pi)
  float x3 = x * x * x;
  return 0.5f * x * (1.0f + tanhf(c * (x + 0.044715f * x3)));
}

// ======================= generic WMMA bf16 GEMM =======================
// C[M,N] = act( A[M,K] @ B[K,N] + bias[N] ), fp32 in/out, bf16 compute.
// Double-buffered pipeline: global loads for tile i+1 issue into registers
// before the barrier; WMMA runs from LDS buffer i; converted stores fill
// buffer i^1. One s_barrier per K-step.
#define TM 64
#define TN 64
#define TK 32
#define TKP (TK + 8)    // 40 elems = 80 B row stride (16B multiple)

template <int ACT, int M, int N, int K, bool NTAIL>
__global__ __launch_bounds__(128)
void gemm_bf16(const float* __restrict__ A, const float* __restrict__ Bw,
               const float* __restrict__ bias, float* __restrict__ C)
{
  __shared__ __align__(16) bf16_t Asm[2][TM][TKP];
  __shared__ __align__(16) bf16_t Bt[2][TN][TKP];

  const int tile_n = blockIdx.x * TN;
  const int tile_m = blockIdx.y * TM;
  const int tid  = threadIdx.x;
  const int lane = tid & 31, wave = tid >> 5;
  const int half = lane >> 4, l16 = lane & 15;
  const int wrow = wave * 16;

  float4 ra[4];
  float4 rb4[4];
  float  rbs[16];

  auto ld_tile = [&](int k0) {
#pragma unroll
    for (int it = 0; it < 4; ++it) {
      int e = (tid + it * 128) << 2;
      int r = e >> 5, c = e & 31;
      ra[it] = *(const float4*)(A + (size_t)(tile_m + r) * K + (k0 + c));
    }
    if constexpr (!NTAIL) {
#pragma unroll
      for (int it = 0; it < 4; ++it) {
        int e = (tid + it * 128) << 2;
        int r = e >> 6, c = e & 63;
        rb4[it] = *(const float4*)(Bw + (size_t)(k0 + r) * N + (tile_n + c));
      }
    } else {
#pragma unroll
      for (int it = 0; it < 16; ++it) {
        int e = tid + it * 128;
        int r = e >> 6, c = e & 63;
        int gc = tile_n + c;
        rbs[it] = (gc < N) ? Bw[(size_t)(k0 + r) * N + gc] : 0.0f;
      }
    }
  };
  auto st_tile = [&](int b) {
#pragma unroll
    for (int it = 0; it < 4; ++it) {
      int e = (tid + it * 128) << 2;
      int r = e >> 5, c = e & 31;
      *(v4bf*)&Asm[b][r][c] = f2bf4(ra[it]);
    }
    if constexpr (!NTAIL) {
#pragma unroll
      for (int it = 0; it < 4; ++it) {
        int e = (tid + it * 128) << 2;
        int r = e >> 6, c = e & 63;
        Bt[b][c + 0][r] = f2bf(rb4[it].x);
        Bt[b][c + 1][r] = f2bf(rb4[it].y);
        Bt[b][c + 2][r] = f2bf(rb4[it].z);
        Bt[b][c + 3][r] = f2bf(rb4[it].w);
      }
    } else {
#pragma unroll
      for (int it = 0; it < 16; ++it) {
        int e = tid + it * 128;
        int r = e >> 6, c = e & 63;
        Bt[b][c][r] = f2bf(rbs[it]);
      }
    }
  };

  v8f acc[4];
#pragma unroll
  for (int n = 0; n < 4; ++n) acc[n] = vzero8();

  constexpr int NKSTEP = K / TK;
  ld_tile(0);
  st_tile(0);
  for (int ki = 0; ki < NKSTEP; ++ki) {
    if (ki + 1 < NKSTEP) ld_tile((ki + 1) * TK);   // prefetch into registers
    __syncthreads();                               // buffer ki&1 visible
    const int b = ki & 1;
    v16bf af = ld_dual8(&Asm[b][wrow + l16][half * 8]);
#pragma unroll
    for (int n = 0; n < 4; ++n) {
      v16bf bfv = ld_dual8(&Bt[b][n * 16 + l16][half * 8]);
      acc[n] = __builtin_amdgcn_wmma_f32_16x16x32_bf16(
          false, af, false, bfv, (short)0, acc[n], false, false);
    }
    if (ki + 1 < NKSTEP) st_tile((ki + 1) & 1);    // fill other buffer
  }

  // epilogue: bias + activation, fp32 store (M % 64 == 0 always)
#pragma unroll
  for (int n = 0; n < 4; ++n) {
    int col = tile_n + n * 16 + l16;
    if (NTAIL && col >= N) continue;
    float bv = bias ? bias[col] : 0.0f;
#pragma unroll
    for (int r = 0; r < 8; ++r) {
      int row = tile_m + wrow + half * 8 + r;
      float vl = acc[n][r] + bv;
      if (ACT == 1) vl = gelu_tanh(vl);
      C[(size_t)row * N + col] = vl;
    }
  }
}

// ======================= BigBird attention (WMMA, online softmax) ==========
// One workgroup = one (head, query block). 4 waves; wave owns 16 query rows.
// DENSE=false: 8 gathered key blocks (sparse path, qb=1..NB-2)
// DENSE=true : all 32 key blocks (global blocks qb=0 and NB-1)
// K/V double-buffered with register prefetch -> one barrier per key block.
// Psm is written and read only by the owning wave (LDS is in-order per wave),
// so the P stash needs no workgroup barrier.
#define HP (HDIM + 8)   // 72 elems = 144 B row stride (16B multiple)
#define BP (BSZ + 8)    // 72 elems = 144 B

template <bool DENSE>
__global__ __launch_bounds__(128)
void attn_kernel(const float* __restrict__ q, const float* __restrict__ k,
                 const float* __restrict__ v, const float* __restrict__ km,
                 const int* __restrict__ bidx, const float* __restrict__ bded,
                 float* __restrict__ ctx)
{
  __shared__ __align__(16) bf16_t Qsm[BSZ][HP];
  __shared__ __align__(16) bf16_t Ksm[2][BSZ][HP];
  __shared__ __align__(16) bf16_t Vt[2][HDIM][BP];
  __shared__ __align__(16) bf16_t Psm[BSZ][BP];

  const int h  = blockIdx.y;
  const int qb = DENSE ? (blockIdx.x ? NBLK - 1 : 0) : ((int)blockIdx.x + 1);
  const int tid  = threadIdx.x;
  const int lane = tid & 31, wave = tid >> 5;
  const int half = lane >> 4, l16 = lane & 15;
  const int wrow = wave * 16;
  const int cbase = h * HDIM;

  float4 rk[8], rv[8];
  auto ld_kv = [&](int blk) {
#pragma unroll
    for (int it = 0; it < 8; ++it) {
      int e = (tid + it * 128) << 2;
      int r = e >> 6, c = e & 63;          // r = key, c = d
      rk[it] = *(const float4*)(k + (size_t)(blk * BSZ + r) * DD + cbase + c);
      rv[it] = *(const float4*)(v + (size_t)(blk * BSZ + r) * DD + cbase + c);
    }
  };
  auto st_kv = [&](int b) {
#pragma unroll
    for (int it = 0; it < 8; ++it) {
      int e = (tid + it * 128) << 2;
      int r = e >> 6, c = e & 63;
      *(v4bf*)&Ksm[b][r][c] = f2bf4(rk[it]);
      Vt[b][c + 0][r] = f2bf(rv[it].x);
      Vt[b][c + 1][r] = f2bf(rv[it].y);
      Vt[b][c + 2][r] = f2bf(rv[it].z);
      Vt[b][c + 3][r] = f2bf(rv[it].w);
    }
  };

  // stage Q block, pre-scaled by 1/sqrt(HD)=0.125
#pragma unroll
  for (int it = 0; it < 8; ++it) {
    int e = (tid + it * 128) << 2;
    int r = e >> 6, c = e & 63;
    float4 fq = *(const float4*)(q + (size_t)(qb * BSZ + r) * DD + cbase + c);
    float4 fs; fs.x = fq.x * 0.125f; fs.y = fq.y * 0.125f;
    fs.z = fq.z * 0.125f; fs.w = fq.w * 0.125f;
    *(v4bf*)&Qsm[r][c] = f2bf4(fs);
  }

  v8f cacc[4];
#pragma unroll
  for (int n = 0; n < 4; ++n) cacc[n] = vzero8();
  float rmax[8], rsum[8];
#pragma unroll
  for (int r = 0; r < 8; ++r) { rmax[r] = -1.0e30f; rsum[r] = 0.0f; }

  const int nkb = DENSE ? NBLK : NKBL;
  int blk0 = DENSE ? 0 : bidx[qb * NKBL + 0];
  ld_kv(blk0);
  st_kv(0);
  int blk = blk0;

  for (int j = 0; j < nkb; ++j) {
    const float ded = DENSE ? 1.0f : bded[qb * NKBL + j];
    const int blk_cur = blk;
    if (j + 1 < nkb) {
      blk = DENSE ? (j + 1) : bidx[qb * NKBL + (j + 1)];
      ld_kv(blk);                          // prefetch next block into registers
    }
    __syncthreads();                       // buffer j&1 visible to all waves
    const int b = j & 1;

    // ---- scores: S = Q (16x64) . K^T (64x64), two K-steps of 32 ----
    v8f sacc[4];
#pragma unroll
    for (int n = 0; n < 4; ++n) sacc[n] = vzero8();
#pragma unroll
    for (int ksX = 0; ksX < 2; ++ksX) {
      v16bf af = ld_dual8(&Qsm[wrow + l16][ksX * 32 + half * 8]);
#pragma unroll
      for (int n = 0; n < 4; ++n) {
        v16bf bfv = ld_dual8(&Ksm[b][n * 16 + l16][ksX * 32 + half * 8]);
        sacc[n] = __builtin_amdgcn_wmma_f32_16x16x32_bf16(
            false, af, false, bfv, (short)0, sacc[n], false, false);
      }
    }

    // mask bias per key column (key = n*16 + l16 within block)
    float mb[4];
#pragma unroll
    for (int n = 0; n < 4; ++n) {
      float mval = km[blk_cur * BSZ + n * 16 + l16] * ded;
      mb[n] = (1.0f - mval) * NEGBIG;
    }

    // ---- online softmax per row (row = half*8 + r within wave strip) ----
    float newmax[8], corr[8];
#pragma unroll
    for (int r = 0; r < 8; ++r) {
      float mx = -3.0e38f;
#pragma unroll
      for (int n = 0; n < 4; ++n) mx = fmaxf(mx, sacc[n][r] + mb[n]);
      mx = redmax16(mx);
      newmax[r] = fmaxf(rmax[r], mx);
      corr[r] = __expf(rmax[r] - newmax[r]);
    }
#pragma unroll
    for (int n = 0; n < 4; ++n)
#pragma unroll
      for (int r = 0; r < 8; ++r)
        sacc[n][r] = __expf(sacc[n][r] + mb[n] - newmax[r]);
#pragma unroll
    for (int r = 0; r < 8; ++r) {
      float sm = 0.0f;
#pragma unroll
      for (int n = 0; n < 4; ++n) sm += sacc[n][r];
      sm = redsum16(sm);
      rsum[r] = rsum[r] * corr[r] + sm;
      rmax[r] = newmax[r];
#pragma unroll
      for (int n = 0; n < 4; ++n) cacc[n][r] = cacc[n][r] * corr[r];
    }

    // stash probabilities as bf16 (per-wave region; LDS in-order per wave)
#pragma unroll
    for (int n = 0; n < 4; ++n)
#pragma unroll
      for (int r = 0; r < 8; ++r)
        Psm[wrow + half * 8 + r][n * 16 + l16] = f2bf(sacc[n][r]);

    // ---- ctx += P (16x64) . V (64x64) ----
#pragma unroll
    for (int ksX = 0; ksX < 2; ++ksX) {
      v16bf af = ld_dual8(&Psm[wrow + l16][ksX * 32 + half * 8]);
#pragma unroll
      for (int n = 0; n < 4; ++n) {
        v16bf bfv = ld_dual8(&Vt[b][n * 16 + l16][ksX * 32 + half * 8]);
        cacc[n] = __builtin_amdgcn_wmma_f32_16x16x32_bf16(
            false, af, false, bfv, (short)0, cacc[n], false, false);
      }
    }

    if (j + 1 < nkb) st_kv((j + 1) & 1);   // fill other buffer
  }

  // normalize and write context (head slice columns)
#pragma unroll
  for (int n = 0; n < 4; ++n)
#pragma unroll
    for (int r = 0; r < 8; ++r) {
      int row = qb * BSZ + wrow + half * 8 + r;
      int col = cbase + n * 16 + l16;
      ctx[(size_t)row * DD + col] = cacc[n][r] / rsum[r];
    }
}

// ======================= residual + LayerNorm over D=768 ====================
__global__ __launch_bounds__(256)
void add_ln_kernel(const float* __restrict__ in, const float* __restrict__ res,
                   const float* __restrict__ sc, const float* __restrict__ bi,
                   float* __restrict__ out)
{
  const int row = blockIdx.x;
  const int c0 = threadIdx.x;
  __shared__ float red[8];

  float vals[3];
#pragma unroll
  for (int i = 0; i < 3; ++i) {
    int c = c0 + i * 256;
    float vl = in[(size_t)row * DD + c];
    if (res) vl += res[(size_t)row * DD + c];
    vals[i] = vl;
  }
  float s = vals[0] + vals[1] + vals[2];
  s = redsum32(s);
  int lane = threadIdx.x & 31, wave = threadIdx.x >> 5;
  if (lane == 0) red[wave] = s;
  __syncthreads();
  float tot = 0.0f;
#pragma unroll
  for (int i = 0; i < 8; ++i) tot += red[i];
  float mean = tot / (float)DD;

  float vs = 0.0f;
#pragma unroll
  for (int i = 0; i < 3; ++i) { float d = vals[i] - mean; vs += d * d; }
  vs = redsum32(vs);
  __syncthreads();
  if (lane == 0) red[wave] = vs;
  __syncthreads();
  float vtot = 0.0f;
#pragma unroll
  for (int i = 0; i < 8; ++i) vtot += red[i];
  float inv = rsqrtf(vtot / (float)DD + 1e-12f);

#pragma unroll
  for (int i = 0; i < 3; ++i) {
    int c = c0 + i * 256;
    out[(size_t)row * DD + c] = (vals[i] - mean) * inv * sc[c] + bi[c];
  }
}

// ======================= embeddings / setup ====================
__global__ __launch_bounds__(256)
void embed_kernel(const int* __restrict__ ids, const int* __restrict__ tt,
                  const float* __restrict__ we, const float* __restrict__ pe,
                  const float* __restrict__ te, float* __restrict__ out)
{
  int s = blockIdx.x;
  int id = ids[s], t = tt[s];
  for (int c = threadIdx.x; c < DD; c += 256)
    out[(size_t)s * DD + c] =
        we[(size_t)id * DD + c] + pe[(size_t)s * DD + c] + te[(size_t)t * DD + c];
}

__global__ void km_kernel(const int* __restrict__ m, float* __restrict__ km) {
  int s = blockIdx.x * blockDim.x + threadIdx.x;
  if (s < SS) km[s] = (float)m[s];
}

// deterministic static block map: [0, nb-1, i-1, i, i+1] + 3 pseudo-random
// distinct blocks; dedup mask keeps first occurrence (matches reference shape).
__global__ void blockmap_kernel(int* __restrict__ bidx, float* __restrict__ bded) {
  int i = threadIdx.x;
  if (i >= NBLK) return;
  int lst[NKBL];
  lst[0] = 0;
  lst[1] = NBLK - 1;
  lst[2] = (i - 1 < 0) ? 0 : i - 1;
  lst[3] = i;
  lst[4] = (i + 1 > NBLK - 1) ? NBLK - 1 : i + 1;
  unsigned rng = 0x9E3779B9u ^ ((unsigned)i * 2654435761u);
  int cnt = 5;
  while (cnt < NKBL) {
    rng = rng * 1664525u + 1013904223u;
    int cand = (int)((rng >> 10) & (NBLK - 1));
    bool used = false;
    for (int t = 0; t < cnt; ++t) if (lst[t] == cand) used = true;
    if (!used) lst[cnt++] = cand;
  }
  for (int t = 0; t < NKBL; ++t) {
    bidx[i * NKBL + t] = lst[t];
    float dd = 1.0f;
    for (int p = 0; p < t; ++p) if (lst[p] == lst[t]) dd = 0.0f;
    bded[i * NKBL + t] = dd;
  }
}

// ======================= orchestration ====================
extern "C" void kernel_launch(void* const* d_in, const int* in_sizes, int n_in,
                              void* d_out, int out_size, void* d_ws, size_t ws_size,
                              hipStream_t stream)
{
  (void)in_sizes; (void)n_in; (void)out_size; (void)ws_size;

  const int*   input_ids = (const int*)d_in[0];
  const int*   attn_mask = (const int*)d_in[1];
  const int*   type_ids  = (const int*)d_in[2];
  const float* word_emb  = (const float*)d_in[3];
  const float* pos_emb   = (const float*)d_in[4];
  const float* type_emb  = (const float*)d_in[5];
  const float* emb_ln_s  = (const float*)d_in[6];
  const float* emb_ln_b  = (const float*)d_in[7];
  const float* Wq = (const float*)d_in[8];
  const float* Wk = (const float*)d_in[9];
  const float* Wv = (const float*)d_in[10];
  const float* bq = (const float*)d_in[11];
  const float* bk = (const float*)d_in[12];
  const float* bv = (const float*)d_in[13];
  const float* Wo = (const float*)d_in[14];
  const float* bo = (const float*)d_in[15];
  const float* ln1s = (const float*)d_in[16];
  const float* ln1b = (const float*)d_in[17];
  const float* W1 = (const float*)d_in[18];
  const float* b1 = (const float*)d_in[19];
  const float* W2 = (const float*)d_in[20];
  const float* b2 = (const float*)d_in[21];
  const float* ln2s = (const float*)d_in[22];
  const float* ln2b = (const float*)d_in[23];
  const float* fc_w = (const float*)d_in[24];
  const float* fc_b = (const float*)d_in[25];

  char* base = (char*)d_ws;
  size_t off = 0;
  auto carve = [&](size_t bytes) -> void* {
    void* p = base + off;
    off += (bytes + 255) & ~(size_t)255;
    return p;
  };
  float* x    = (float*)carve(sizeof(float) * SS * DD);
  float* tmp  = (float*)carve(sizeof(float) * SS * DD);
  float* qbuf = (float*)carve(sizeof(float) * SS * DD);
  float* kbuf = (float*)carve(sizeof(float) * SS * DD);
  float* vbuf = (float*)carve(sizeof(float) * SS * DD);
  float* ctx  = (float*)carve(sizeof(float) * SS * DD);
  float* hbuf = (float*)carve(sizeof(float) * SS * FF);
  int*   bidx = (int*)carve(sizeof(int) * NBLK * NKBL);
  float* bded = (float*)carve(sizeof(float) * NBLK * NKBL);
  float* kmb  = (float*)carve(sizeof(float) * SS);

  blockmap_kernel<<<1, NBLK, 0, stream>>>(bidx, bded);
  km_kernel<<<(SS + 255) / 256, 256, 0, stream>>>(attn_mask, kmb);
  embed_kernel<<<SS, 256, 0, stream>>>(input_ids, type_ids, word_emb, pos_emb,
                                       type_emb, tmp);
  add_ln_kernel<<<SS, 256, 0, stream>>>(tmp, nullptr, emb_ln_s, emb_ln_b, x);

  const dim3 blk128(128);
  const dim3 gD(DD / 64, SS / 64);
  const dim3 gF(FF / 64, SS / 64);
  const dim3 gV((VV + 63) / 64, SS / 64);

  for (int l = 0; l < LL; ++l) {
    const float* Wq_l = Wq + (size_t)l * DD * DD;
    const float* Wk_l = Wk + (size_t)l * DD * DD;
    const float* Wv_l = Wv + (size_t)l * DD * DD;
    const float* Wo_l = Wo + (size_t)l * DD * DD;
    const float* bq_l = bq + (size_t)l * DD;
    const float* bk_l = bk + (size_t)l * DD;
    const float* bv_l = bv + (size_t)l * DD;
    const float* bo_l = bo + (size_t)l * DD;
    const float* W1_l = W1 + (size_t)l * DD * FF;
    const float* b1_l = b1 + (size_t)l * FF;
    const float* W2_l = W2 + (size_t)l * FF * DD;
    const float* b2_l = b2 + (size_t)l * DD;

    gemm_bf16<0, SS, DD, DD, false><<<gD, blk128, 0, stream>>>(x, Wq_l, bq_l, qbuf);
    gemm_bf16<0, SS, DD, DD, false><<<gD, blk128, 0, stream>>>(x, Wk_l, bk_l, kbuf);
    gemm_bf16<0, SS, DD, DD, false><<<gD, blk128, 0, stream>>>(x, Wv_l, bv_l, vbuf);

    attn_kernel<false><<<dim3(NBLK - 2, HH), blk128, 0, stream>>>(
        qbuf, kbuf, vbuf, kmb, bidx, bded, ctx);
    attn_kernel<true><<<dim3(2, HH), blk128, 0, stream>>>(
        qbuf, kbuf, vbuf, kmb, bidx, bded, ctx);

    gemm_bf16<0, SS, DD, DD, false><<<gD, blk128, 0, stream>>>(ctx, Wo_l, bo_l, tmp);
    add_ln_kernel<<<SS, 256, 0, stream>>>(tmp, x, ln1s + (size_t)l * DD,
                                          ln1b + (size_t)l * DD, x);

    gemm_bf16<1, SS, FF, DD, false><<<gF, blk128, 0, stream>>>(x, W1_l, b1_l, hbuf);
    gemm_bf16<0, SS, DD, FF, false><<<gD, blk128, 0, stream>>>(hbuf, W2_l, b2_l, tmp);
    add_ln_kernel<<<SS, 256, 0, stream>>>(tmp, x, ln2s + (size_t)l * DD,
                                          ln2b + (size_t)l * DD, x);
  }

  // LM head: [2048,768] x [768,50358] -> logits fp32 (N-tail guarded)
  gemm_bf16<0, SS, VV, DD, true><<<gV, blk128, 0, stream>>>(x, fc_w, fc_b,
                                                            (float*)d_out);
}
